// Decoder_basesup3_6528350290200
// MI455X (gfx1250) — matrix operand order repaired and checked by reading, not compiled, not run on hardware
//
#include <hip/hip_runtime.h>
#include <hip/hip_bf16.h>
#include <stdint.h>

// ---------------------------------------------------------------------------
// MI455X (gfx1250, wave32) mesh-decoder.
// GEMMs: v_wmma_f32_16x16x32_f16 (f16 in, f32 accum), K-loop fully unrolled
// via compile-time CPAD/NOUT. Graph gathers (adj-mean, unpool repeat,
// face->node vertex mean) are fused into the WMMA A-fragment loader using
// packed v_pk_add_f16 math. Weights are repacked fragment-major so the B
// operand is a single 32B vector load. Scatter uses global_atomic_add_f32.
// ---------------------------------------------------------------------------

typedef __attribute__((ext_vector_type(16))) _Float16 v16h;
typedef __attribute__((ext_vector_type(8)))  _Float16 v8h;
typedef __attribute__((ext_vector_type(8)))  float    v8f;

static __device__ __forceinline__ v8h ldv8(const _Float16* p) {
    return *(const v8h*)p;    // 16B vector load (rows 64/128B-strided, 256B-aligned base)
}
static __device__ __forceinline__ v16h ldv16(const _Float16* p) {
    return *(const v16h*)p;   // 32B vector load (fragment-major packed weights)
}

// ---------------------------------------------------------------------------
// Fused gather + WMMA GEMM.
//   MODE 0 ("fo"):   A[m] = mean(src[tri[m][0..2]])                      KTOT = CPAD
//   MODE 1 ("conv"): A[m] = concat[src[m>>sh], mean(src[tri[m]>>sh])]    KTOT = 2*CPAD
// W is pre-packed f16 fragment-major:
//   W[((kc/32)*NT + nt)*512 + lane*16 + j] = Wpad[kc + (lane>>4)*16 + j][nt*16 + (lane&15)]
// with zero fill in padded K rows (so garbage in padded A columns is killed).
// One wave = one 16x16 D tile; all early-exits are wave-uniform (EXEC all-1s).
// ---------------------------------------------------------------------------
template<int MODE, bool RELU, int CPAD, int NOUT>
__global__ void gemm_gather_kernel(const _Float16* __restrict__ src, int srcShift,
                                   const int* __restrict__ tri,
                                   const _Float16* __restrict__ W,
                                   const float* __restrict__ bias,
                                   _Float16* __restrict__ out,
                                   int F)
{
    constexpr int NT   = NOUT / 16;
    constexpr int KTOT = (MODE == 1) ? (2 * CPAD) : CPAD;

    const int wid = (int)((blockIdx.x * blockDim.x + threadIdx.x) >> 5);
    const int mt  = wid / NT;
    const int nt  = wid - mt * NT;
    if (mt * 16 >= F) return;                 // wave-uniform

    const int lane = threadIdx.x & 31;
    const int half = lane >> 4;               // A: K-half select
    const int mr   = lane & 15;               // A: row in tile; B/D: column
    const int row  = mt * 16 + mr;

    const _Float16* arow = src + ((size_t)(row >> srcShift)) * (size_t)CPAD;
    const _Float16* g0   = src + ((size_t)(tri[row*3+0] >> srcShift)) * (size_t)CPAD;
    const _Float16* g1   = src + ((size_t)(tri[row*3+1] >> srcShift)) * (size_t)CPAD;
    const _Float16* g2   = src + ((size_t)(tri[row*3+2] >> srcShift)) * (size_t)CPAD;

    const _Float16 th = (_Float16)(1.0f / 3.0f);
    const v8h thv = {th, th, th, th, th, th, th, th};

    v8f acc = {0.f,0.f,0.f,0.f,0.f,0.f,0.f,0.f};

    #pragma unroll
    for (int kc = 0; kc < KTOT; kc += 32) {
        const bool ident = (MODE == 1) && (kc < CPAD);       // constant after unroll
        const int  kl    = ident ? kc : (kc - ((MODE == 1) ? CPAD : 0));
        const int  o0    = kl + half * 16;                   // A lanes: K = half*16 + [0..7]
        const int  o1    = kl + half * 16 + 8;               //          K = half*16 + 8 + [0..7]

        v8h h0, h1;
        if (ident) {
            h0 = ldv8(arow + o0);
            h1 = ldv8(arow + o1);
        } else {
            // 3-row mean in packed f16 (v_pk_add_f16 / v_pk_mul_f16)
            h0 = (ldv8(g0 + o0) + ldv8(g1 + o0) + ldv8(g2 + o0)) * thv;
            h1 = (ldv8(g0 + o1) + ldv8(g1 + o1) + ldv8(g2 + o1)) * thv;
        }
        v16h a;
        #pragma unroll
        for (int j = 0; j < 8; ++j) { a[j] = h0[j]; a[8+j] = h1[j]; }

        // fragment-major B: one contiguous 32B load per lane
        v16h b = ldv16(W + (((size_t)(kc >> 5)) * NT + nt) * 512 + lane * 16);

        acc = __builtin_amdgcn_wmma_f32_16x16x32_f16(
                  false, a, false, b, (short)0, acc, false, false);
    }

    const float bv = bias[nt * 16 + mr];
    #pragma unroll
    for (int r = 0; r < 8; ++r) {
        float v = acc[r] + bv;
        if (RELU) v = fmaxf(v, 0.0f);
        // D layout: lane half 0 -> rows 0..7, half 1 -> rows 8..15; col = mr
        out[(size_t)(mt*16 + half*8 + r) * NOUT + (nt*16 + mr)] = (_Float16)v;
    }
}

template<int MODE, bool RELU, int CPAD, int NOUT>
static void launch_gemm(const _Float16* src, int shift, const int* tri, const _Float16* W,
                        const float* bias, _Float16* out, int F, hipStream_t stream)
{
    long tiles = (long)(F / 16) * (NOUT / 16);
    gemm_gather_kernel<MODE, RELU, CPAD, NOUT>
        <<<dim3((unsigned)((tiles + 7) / 8)), 256, 0, stream>>>(
            src, shift, tri, W, bias, out, F);
}

// ---------------------------------------------------------------------------
// Weight repack: f32 [Krows x N] -> f16 fragment-major, zero-filled padding.
// Logical padded matrix Wpad[Ktot x N]: rows [0,C) = W[0..C);
// parts==2: rows [Cpad, Cpad+C) = W[C..2C); all other rows zero.
// ---------------------------------------------------------------------------
__global__ void pack_weight_kernel(const float* __restrict__ W, _Float16* __restrict__ dst,
                                   int C, int Cpad, int parts, int N, int Ktot)
{
    int i = blockIdx.x * blockDim.x + threadIdx.x;
    if (i >= Ktot * N) return;
    int j    = i & 15;
    int lane = (i >> 4) & 31;
    int rest = i >> 9;
    int NT   = N >> 4;
    int nt   = rest % NT;
    int chunk = rest / NT;
    int k = chunk * 32 + (lane >> 4) * 16 + j;
    int n = nt * 16 + (lane & 15);
    float v = 0.f;
    if (k < C)                                        v = W[k * N + n];
    else if (parts == 2 && k >= Cpad && k < Cpad + C) v = W[(k - Cpad + C) * N + n];
    dst[i] = (_Float16)v;
}

// ---------------------------------------------------------------------------
// Face geometric features (normal, centroid, half-area) + input features,
// packed f16, row stride 32 (cols 23..31 zeroed).
// ---------------------------------------------------------------------------
__global__ void extract_features_kernel(const float* __restrict__ pos, const int* __restrict__ faces,
                                        const float* __restrict__ feat, _Float16* __restrict__ out, int F)
{
    int f = blockIdx.x * blockDim.x + threadIdx.x;
    if (f >= F) return;
    int ia = faces[f*3+0], ib = faces[f*3+1], ic = faces[f*3+2];
    float ax = pos[ia*3+0], ay = pos[ia*3+1], az = pos[ia*3+2];
    float bx = pos[ib*3+0], by = pos[ib*3+1], bz = pos[ib*3+2];
    float cx = pos[ic*3+0], cy = pos[ic*3+1], cz = pos[ic*3+2];
    float e1x = bx-ax, e1y = by-ay, e1z = bz-az;
    float e2x = cx-ax, e2y = cy-ay, e2z = cz-az;
    float nx = e1y*e2z - e1z*e2y;
    float ny = e1z*e2x - e1x*e2z;
    float nz = e1x*e2y - e1y*e2x;
    float l   = sqrtf(nx*nx + ny*ny + nz*nz);
    float inv = 1.0f / fmaxf(l, 1e-12f);
    _Float16* o = out + (size_t)f * 32;
    o[0] = (_Float16)(nx*inv); o[1] = (_Float16)(ny*inv); o[2] = (_Float16)(nz*inv);
    o[3] = (_Float16)((ax+bx+cx)*(1.0f/3.0f));
    o[4] = (_Float16)((ay+by+cy)*(1.0f/3.0f));
    o[5] = (_Float16)((az+bz+cz)*(1.0f/3.0f));
    o[6] = (_Float16)(0.5f*l);
    #pragma unroll
    for (int j = 0; j < 16; ++j) o[7+j] = (_Float16)feat[f*16+j];
    #pragma unroll
    for (int j = 23; j < 32; ++j) o[j] = (_Float16)0.0f;
}

__global__ void zero_f32_kernel(float* p, int n)
{
    int i = blockIdx.x * blockDim.x + threadIdx.x;
    if (i < n) p[i] = 0.f;
}

// face2node scatter: nf[v] += ff[f]; cnt[v] += 1  (global_atomic_add_f32)
__global__ void scatter_faces_kernel(const _Float16* __restrict__ ff, const int* __restrict__ faces,
                                     float* __restrict__ nf, float* __restrict__ cnt, int F, int C)
{
    int t = blockIdx.x * blockDim.x + threadIdx.x;
    if (t >= F * C) return;
    int f = t / C, c = t - f * C;
    float v = (float)ff[(size_t)t];
    int v0 = faces[f*3+0], v1 = faces[f*3+1], v2 = faces[f*3+2];
    atomicAdd(&nf[(size_t)v0 * C + c], v);
    atomicAdd(&nf[(size_t)v1 * C + c], v);
    atomicAdd(&nf[(size_t)v2 * C + c], v);
    if (c == 0) {
        atomicAdd(&cnt[v0], 1.f);
        atomicAdd(&cnt[v1], 1.f);
        atomicAdd(&cnt[v2], 1.f);
    }
}

// normalize nf by count, emit f16 node features, pos += nf @ Wp + bp
__global__ void node_update_kernel(const float* __restrict__ nf, const float* __restrict__ cnt,
                                   const float* __restrict__ Wp, const float* __restrict__ bp,
                                   float* __restrict__ pos, _Float16* __restrict__ nfh,
                                   int Nn, int C)
{
    int n = blockIdx.x * blockDim.x + threadIdx.x;
    if (n >= Nn) return;
    float inv = 1.0f / fmaxf(cnt[n], 1.0f);
    float s0 = bp[0], s1 = bp[1], s2 = bp[2];
    const float* r = nf + (size_t)n * C;
    for (int c = 0; c < C; ++c) {
        float v = r[c] * inv;
        if (nfh) nfh[(size_t)n * C + c] = (_Float16)v;
        s0 += v * Wp[c*3+0];
        s1 += v * Wp[c*3+1];
        s2 += v * Wp[c*3+2];
    }
    pos[n*3+0] += s0; pos[n*3+1] += s1; pos[n*3+2] += s2;
}

// unpool: append edge midpoints (edge endpoints always < baseN -> no race)
__global__ void unpool_pos_kernel(const int* __restrict__ edges, int E, int baseN,
                                  float* __restrict__ pos)
{
    int e = blockIdx.x * blockDim.x + threadIdx.x;
    if (e >= E) return;
    int a = edges[e*2+0], b = edges[e*2+1];
    #pragma unroll
    for (int k = 0; k < 3; ++k)
        pos[(size_t)(baseN + e)*3 + k] = 0.5f * (pos[a*3+k] + pos[b*3+k]);
}

// ff48 = concat[fo3 (F0 x 32), input_feature (F0 x 16)], row stride 64, zero pad
__global__ void build_ff48_kernel(const _Float16* __restrict__ fo, const float* __restrict__ feat,
                                  _Float16* __restrict__ out, int F)
{
    int t = blockIdx.x * blockDim.x + threadIdx.x;
    if (t >= F * 64) return;
    int f = t >> 6, c = t & 63;
    _Float16 v = (_Float16)0.0f;
    if      (c < 32) v = fo[(size_t)f*32 + c];
    else if (c < 48) v = (_Float16)feat[f*16 + (c - 32)];
    out[(size_t)t] = v;
}

__global__ void copy_f32_kernel(float* __restrict__ dst, const float* __restrict__ src, int n)
{
    int i = blockIdx.x * blockDim.x + threadIdx.x;
    if (i < n) dst[i] = src[i];
}

__global__ void i32_to_f32_kernel(float* __restrict__ dst, const int* __restrict__ src, int n)
{
    int i = blockIdx.x * blockDim.x + threadIdx.x;
    if (i < n) dst[i] = (float)src[i];
}

// ---------------------------------------------------------------------------
extern "C" void kernel_launch(void* const* d_in, const int* in_sizes, int n_in,
                              void* d_out, int out_size, void* d_ws, size_t ws_size,
                              hipStream_t stream)
{
    // setup_inputs() flat order (params dict in insertion order, tuple members in order)
    enum {
        IN_POS = 0, IN_FEAT = 1,
        P_CONV1S_W, P_CONV1S_B,
        P_F2N1S_WP, P_F2N1S_BP, P_F2N1S_WF, P_F2N1S_BF,
        P_CONV2S_W, P_CONV2S_B,
        P_F2N2S_WP, P_F2N2S_BP, P_F2N2S_WF, P_F2N2S_BF,
        P_CONV3S_W, P_CONV3S_B,
        P_F2N3S_WP, P_F2N3S_BP, P_F2N3S_WF, P_F2N3S_BF,
        P_CONV1_W, P_CONV1_B,
        P_F2N1_WP, P_F2N1_BP, P_F2N1_WF, P_F2N1_BF,
        P_CONV2_W, P_CONV2_B,
        P_F2N2_WP, P_F2N2_BP, P_F2N2_WF, P_F2N2_BF,
        P_CONV3_W, P_CONV3_B,
        P_F2N3_WP, P_F2N3_BP,
        IN_FACES0, IN_FACES1, IN_FACES2, IN_FACES3,
        IN_E0, IN_E1, IN_E2,
        IN_ADJ0, IN_ADJ1, IN_ADJ2, IN_ADJ3
    };

    const float* posIn = (const float*)d_in[IN_POS];
    const float* feat  = (const float*)d_in[IN_FEAT];
    const int* faces0 = (const int*)d_in[IN_FACES0];
    const int* faces1 = (const int*)d_in[IN_FACES1];
    const int* faces2 = (const int*)d_in[IN_FACES2];
    const int* faces3 = (const int*)d_in[IN_FACES3];
    const int* edges0 = (const int*)d_in[IN_E0];
    const int* edges1 = (const int*)d_in[IN_E1];
    const int* edges2 = (const int*)d_in[IN_E2];
    const int* adj0 = (const int*)d_in[IN_ADJ0];
    const int* adj1 = (const int*)d_in[IN_ADJ1];
    const int* adj2 = (const int*)d_in[IN_ADJ2];
    const int* adj3 = (const int*)d_in[IN_ADJ3];

    const int F0 = in_sizes[IN_FACES0] / 3, F1 = in_sizes[IN_FACES1] / 3;
    const int F2 = in_sizes[IN_FACES2] / 3, F3 = in_sizes[IN_FACES3] / 3;
    const int N0 = in_sizes[IN_POS] / 3;
    const int E0 = in_sizes[IN_E0] / 2, E1 = in_sizes[IN_E1] / 2, E2 = in_sizes[IN_E2] / 2;
    const int N1 = N0 + E0, N2 = N1 + E1, N3 = N2 + E2;

    // ---- workspace carve-out (bump allocator, 256B aligned) ----
    char* wsp = (char*)d_ws; size_t off = 0;
    auto alloc = [&](size_t bytes) -> void* {
        void* p = wsp + off;
        off = (off + bytes + 255) & ~(size_t)255;
        return p;
    };
    auto allocH = [&](size_t halfs) { return (_Float16*)alloc(halfs * sizeof(_Float16)); };
    auto allocF = [&](size_t n)     { return (float*)alloc(n * sizeof(float)); };

    _Float16* wk_c1s  = allocH(64 * 64);    // conv1s: C=23 Cpad=32 Ktot=64  N=64
    _Float16* wk_f1s  = allocH(64 * 64);    // f2n1s Wf: 64x64
    _Float16* wk_c2s  = allocH(128 * 64);   // conv2s: C=64 Ktot=128 N=64
    _Float16* wk_f2s  = allocH(64 * 64);
    _Float16* wk_c3s  = allocH(128 * 64);
    _Float16* wk_f3s  = allocH(64 * 32);    // f2n3s Wf: 64x32
    _Float16* wk_c1   = allocH(128 * 32);   // conv1: C=48 Cpad=64 Ktot=128 N=32
    _Float16* wk_f1   = allocH(32 * 32);
    _Float16* wk_c2   = allocH(64 * 32);    // conv2/3: C=32 Ktot=64 N=32
    _Float16* wk_f2   = allocH(32 * 32);
    _Float16* wk_c3   = allocH(64 * 32);

    _Float16* ff0  = allocH((size_t)F0 * 32);
    _Float16* bufA = allocH((size_t)F0 * 64);
    _Float16* bufB = allocH((size_t)F0 * 64);
    _Float16* ff48 = allocH((size_t)F0 * 64);
    _Float16* ffc1 = allocH((size_t)F1 * 32);
    _Float16* fo1  = allocH((size_t)F1 * 32);
    _Float16* ffc2 = allocH((size_t)F2 * 32);
    _Float16* fo2  = allocH((size_t)F2 * 32);
    _Float16* ffc3 = allocH((size_t)F3 * 32);

    size_t maxNF = (size_t)N0 * 64;
    if ((size_t)N1 * 32 > maxNF) maxNF = (size_t)N1 * 32;
    if ((size_t)N2 * 32 > maxNF) maxNF = (size_t)N2 * 32;
    if ((size_t)N3 * 32 > maxNF) maxNF = (size_t)N3 * 32;
    float*    nf   = allocF(maxNF);
    _Float16* nfh  = allocH(maxNF);
    float*    cnt  = allocF((size_t)N3);
    float*    posb = allocF((size_t)N3 * 3);
    (void)ws_size; (void)n_in; (void)out_size;

    auto blocks = [](long n) { return dim3((unsigned)((n + 255) / 256)); };

    auto pack = [&](int idxW, _Float16* dst, int C, int Cpad, int parts, int N, int Ktot) {
        pack_weight_kernel<<<blocks((long)Ktot * N), 256, 0, stream>>>(
            (const float*)d_in[idxW], dst, C, Cpad, parts, N, Ktot);
    };
    auto f2n = [&](const _Float16* ff, const int* faces, int idxWp, int idxBp,
                   _Float16* nfhOut, int F, int Nn, int C) {
        zero_f32_kernel<<<blocks((long)Nn * C), 256, 0, stream>>>(nf, Nn * C);
        zero_f32_kernel<<<blocks(Nn), 256, 0, stream>>>(cnt, Nn);
        scatter_faces_kernel<<<blocks((long)F * C), 256, 0, stream>>>(ff, faces, nf, cnt, F, C);
        node_update_kernel<<<blocks(Nn), 256, 0, stream>>>(
            nf, cnt, (const float*)d_in[idxWp], (const float*)d_in[idxBp],
            posb, nfhOut, Nn, C);
    };
    auto bias = [&](int idx) { return (const float*)d_in[idx]; };

    // ---- pack weights (tiny, L1/L2-resident during GEMMs) ----
    pack(P_CONV1S_W, wk_c1s, 23, 32, 2, 64, 64);
    pack(P_F2N1S_WF, wk_f1s, 64, 64, 1, 64, 64);
    pack(P_CONV2S_W, wk_c2s, 64, 64, 2, 64, 128);
    pack(P_F2N2S_WF, wk_f2s, 64, 64, 1, 64, 64);
    pack(P_CONV3S_W, wk_c3s, 64, 64, 2, 64, 128);
    pack(P_F2N3S_WF, wk_f3s, 64, 64, 1, 32, 64);
    pack(P_CONV1_W,  wk_c1,  48, 64, 2, 32, 128);
    pack(P_F2N1_WF,  wk_f1,  32, 32, 1, 32, 32);
    pack(P_CONV2_W,  wk_c2,  32, 32, 2, 32, 64);
    pack(P_F2N2_WF,  wk_f2,  32, 32, 1, 32, 32);
    pack(P_CONV3_W,  wk_c3,  32, 32, 2, 32, 64);

    // ---- level 0 (coarse mesh, HID=64) ----
    copy_f32_kernel<<<blocks((long)N0 * 3), 256, 0, stream>>>(posb, posIn, N0 * 3);
    extract_features_kernel<<<blocks(F0), 256, 0, stream>>>(posb, faces0, feat, ff0, F0);

    launch_gemm<1,true, 32,64>(ff0,  0, adj0, wk_c1s, bias(P_CONV1S_B), bufA, F0, stream);
    f2n(bufA, faces0, P_F2N1S_WP, P_F2N1S_BP, nfh, F0, N0, 64);
    launch_gemm<0,false,64,64>(nfh,  0, faces0, wk_f1s, bias(P_F2N1S_BF), bufB, F0, stream);

    launch_gemm<1,true, 64,64>(bufB, 0, adj0, wk_c2s, bias(P_CONV2S_B), bufA, F0, stream);
    f2n(bufA, faces0, P_F2N2S_WP, P_F2N2S_BP, nfh, F0, N0, 64);
    launch_gemm<0,false,64,64>(nfh,  0, faces0, wk_f2s, bias(P_F2N2S_BF), bufB, F0, stream);

    launch_gemm<1,true, 64,64>(bufB, 0, adj0, wk_c3s, bias(P_CONV3S_B), bufA, F0, stream);
    f2n(bufA, faces0, P_F2N3S_WP, P_F2N3S_BP, nfh, F0, N0, 64);
    launch_gemm<0,false,64,32>(nfh,  0, faces0, wk_f3s, bias(P_F2N3S_BF), bufB, F0, stream);  // fo3: F0 x 32

    // pos_base -> output slot 0 (before finer levels touch rows [0,N0))
    float* outp = (float*)d_out;
    copy_f32_kernel<<<blocks((long)N0 * 3), 256, 0, stream>>>(outp, posb, N0 * 3);

    // ---- level 1 (HID2=32) : unpool handled by srcShift=2 (repeat(ff,4)) ----
    build_ff48_kernel<<<blocks((long)F0 * 64), 256, 0, stream>>>(bufB, feat, ff48, F0);
    unpool_pos_kernel<<<blocks(E0), 256, 0, stream>>>(edges0, E0, N0, posb);
    launch_gemm<1,true, 64,32>(ff48, 2, adj1, wk_c1, bias(P_CONV1_B), ffc1, F1, stream);
    f2n(ffc1, faces1, P_F2N1_WP, P_F2N1_BP, nfh, F1, N1, 32);
    launch_gemm<0,false,32,32>(nfh,  0, faces1, wk_f1, bias(P_F2N1_BF), fo1, F1, stream);

    // ---- level 2 ----
    unpool_pos_kernel<<<blocks(E1), 256, 0, stream>>>(edges1, E1, N1, posb);
    launch_gemm<1,true, 32,32>(fo1,  2, adj2, wk_c2, bias(P_CONV2_B), ffc2, F2, stream);
    f2n(ffc2, faces2, P_F2N2_WP, P_F2N2_BP, nfh, F2, N2, 32);
    launch_gemm<0,false,32,32>(nfh,  0, faces2, wk_f2, bias(P_F2N2_BF), fo2, F2, stream);

    // ---- level 3 ----
    unpool_pos_kernel<<<blocks(E2), 256, 0, stream>>>(edges2, E2, N2, posb);
    launch_gemm<1,true, 32,32>(fo2,  2, adj3, wk_c3, bias(P_CONV3_B), ffc3, F3, stream);
    f2n(ffc3, faces3, P_F2N3_WP, P_F2N3_BP, (_Float16*)nullptr, F3, N3, 32);

    // ---- assemble outputs: pos_base | pos_fine | faces | faces3 ----
    long o = (long)N0 * 3;
    copy_f32_kernel<<<blocks((long)N3 * 3), 256, 0, stream>>>(outp + o, posb, N3 * 3);
    o += (long)N3 * 3;
    i32_to_f32_kernel<<<blocks((long)F0 * 3), 256, 0, stream>>>(outp + o, faces0, F0 * 3);
    o += (long)F0 * 3;
    i32_to_f32_kernel<<<blocks((long)F3 * 3), 256, 0, stream>>>(outp + o, faces3, F3 * 3);
}
